// MHSA_21973052686444
// MI455X (gfx1250) — compile-verified, matrix operand order checked by reading, not run on hardware
//
#include <hip/hip_runtime.h>

// ---------------------------------------------------------------------------
// MHSA forward for gfx1250 (MI455X): bf16 WMMA everywhere, flash-attention
// style online softmax, register-blocked GEMMs (32x64 per wave).
// B=2, S=2048, H=1024, heads=16, d=64.
// ---------------------------------------------------------------------------

typedef __bf16 bf16;
typedef bf16 v16bf __attribute__((ext_vector_type(16)));
typedef bf16 v8bf  __attribute__((ext_vector_type(8)));
typedef float v8f  __attribute__((ext_vector_type(8)));

#define WMMA_BF16(a, b, c) \
  __builtin_amdgcn_wmma_f32_16x16x32_bf16(false, (a), false, (b), (short)0, (c), false, false)

// Load a 16x32 bf16 operand fragment (A-operand layout; B-operand symmetric,
// fed from K-contiguous rows = columns of the K x 16 matrix).
// Lanes 0-15 hold K={0..7,16..23}, lanes 16-31 hold K={8..15,24..31}, row=lane%16.
__device__ __forceinline__ v16bf load_frag16(const bf16* base, int ld, int k0) {
  const int lane = threadIdx.x & 31;
  const int row  = lane & 15;
  const int hi   = (lane >> 4) << 3;
  const bf16* p  = base + (size_t)row * ld + k0 + hi;
  union { v16bf f; v8bf h[2]; } u;
  u.h[0] = *(const v8bf*)(p);
  u.h[1] = *(const v8bf*)(p + 16);
  return u.f;
}

// ---------------------------------------------------------------------------
// Kernel 1: f32 -> bf16 cast, 8 elements per thread (b128 in, b128 out)
// ---------------------------------------------------------------------------
__global__ void __launch_bounds__(256) cast_f32_bf16(const float* __restrict__ src,
                                                     bf16* __restrict__ dst) {
  const size_t i = ((size_t)blockIdx.x * 256 + threadIdx.x) * 8;
  const float4 f0 = *(const float4*)(src + i);
  const float4 f1 = *(const float4*)(src + i + 4);
  v8bf o;
  o[0] = (bf16)f0.x; o[1] = (bf16)f0.y; o[2] = (bf16)f0.z; o[3] = (bf16)f0.w;
  o[4] = (bf16)f1.x; o[5] = (bf16)f1.y; o[6] = (bf16)f1.z; o[7] = (bf16)f1.w;
  *(v8bf*)(dst + i) = o;
}

// ---------------------------------------------------------------------------
// Kernel 2: QKV projection. C[4096,3072] = Xb @ Wqkv^T / 32.
// One wave computes a 32x64 block (2 A-frags x 4 B-frags -> 8 WMMAs / k-step).
// Scatter into Q[b,h,s,d], K[b,h,s,d], and V^T[b,h,d,s] (bf16).
// The 64-col block is 64-aligned => z (q/k/v) and head are wave-uniform.
// ---------------------------------------------------------------------------
__global__ void __launch_bounds__(256) gemm_qkv_kernel(const bf16* __restrict__ xb,
                                                       const bf16* __restrict__ wb,
                                                       bf16* __restrict__ qb,
                                                       bf16* __restrict__ kb,
                                                       bf16* __restrict__ vtb) {
  const int wave = (blockIdx.x * 256 + threadIdx.x) >> 5;
  const int mt = wave / 48;                         // 0..127 (32 rows each)
  const int nt = wave % 48;                         // 0..47  (64 cols each)
  const bf16* A0 = xb + (size_t)mt * 32 * 1024;
  const bf16* A1 = A0 + 16 * 1024;
  const bf16* B  = wb + (size_t)nt * 64 * 1024;
  v8f acc[2][4] = {};
  for (int kk = 0; kk < 1024; kk += 32) {
    const v16bf a0 = load_frag16(A0, 1024, kk);
    const v16bf a1 = load_frag16(A1, 1024, kk);
#pragma unroll
    for (int j = 0; j < 4; ++j) {
      const v16bf bfr = load_frag16(B + (size_t)j * 16 * 1024, 1024, kk);
      acc[0][j] = WMMA_BF16(a0, bfr, acc[0][j]);
      acc[1][j] = WMMA_BF16(a1, bfr, acc[1][j]);
    }
  }
  const int lane  = threadIdx.x & 31;
  const int nbase = nt * 64;
  const int z  = nbase >> 10;                       // 0=q 1=k 2=v (uniform)
  const int hh = (nbase >> 6) & 15;                 // head (uniform)
  const int m0 = mt * 32 + ((lane >> 4) << 3);
#pragma unroll
  for (int t = 0; t < 2; ++t) {
#pragma unroll
    for (int j = 0; j < 4; ++j) {
      const int dd = j * 16 + (lane & 15);
#pragma unroll
      for (int r = 0; r < 8; ++r) {
        const int m  = m0 + t * 16 + r;
        const int bb = m >> 11;                     // batch
        const int s  = m & 2047;                    // sequence pos
        const bf16 val = (bf16)(acc[t][j][r] * 0.03125f);   // * 1024^-0.5
        const size_t bh = (size_t)(bb * 16 + hh);
        if (z == 0)      qb[(bh * 2048 + s) * 64 + dd] = val;
        else if (z == 1) kb[(bh * 2048 + s) * 64 + dd] = val;
        else             vtb[(bh * 64 + dd) * 2048 + s] = val;
      }
    }
  }
}

// ---------------------------------------------------------------------------
// Kernel 3: flash attention. One wave = one (b,h,16-query) tile.
// Loops over keys in 32-key steps; online softmax; 16x64 f32 accumulator.
// ---------------------------------------------------------------------------
__global__ void __launch_bounds__(256) attn_kernel(const bf16* __restrict__ qbuf,
                                                   const bf16* __restrict__ kbuf,
                                                   const bf16* __restrict__ vtb,
                                                   bf16* __restrict__ ob) {
  __shared__ __attribute__((aligned(16))) bf16 lds[8][16 * 32];  // 1KB per wave
  const int lane = threadIdx.x & 31;
  const int wl   = threadIdx.x >> 5;
  const int wave = blockIdx.x * 8 + wl;
  const int bh = wave >> 7;                          // 0..31
  const int qt = wave & 127;                         // query tile 0..127
  const int b  = bh >> 4;
  const int h  = bh & 15;
  const bf16* Q = qbuf + ((size_t)bh * 2048 + qt * 16) * 64;
  const bf16* K = kbuf + (size_t)bh * 2048 * 64;
  const bf16* V = vtb  + (size_t)bh * 64 * 2048;     // [d][s]

  const v16bf qa0 = load_frag16(Q, 64, 0);           // d = 0..31
  const v16bf qa1 = load_frag16(Q, 64, 32);          // d = 32..63

  v8f o0 = {}, o1 = {}, o2 = {}, o3 = {};            // 16x64 output accum
  float mrun[8], lrun[8];
#pragma unroll
  for (int r = 0; r < 8; ++r) { mrun[r] = -1e30f; lrun[r] = 0.0f; }

  const int colbase = lane & 15;
  const int rowoff  = (lane >> 4) << 3;

  for (int kt = 0; kt < 2048; kt += 32) {
    // ---- scores S[16q x 32k] = Q @ K^T * d^-0.5 ----
    v8f s0 = {}, s1 = {};
    {
      v16bf t = load_frag16(K + (size_t)kt * 64, 64, 0);
      s0 = WMMA_BF16(qa0, t, s0);
      t = load_frag16(K + (size_t)kt * 64, 64, 32);
      s0 = WMMA_BF16(qa1, t, s0);
      t = load_frag16(K + (size_t)(kt + 16) * 64, 64, 0);
      s1 = WMMA_BF16(qa0, t, s1);
      t = load_frag16(K + (size_t)(kt + 16) * 64, 64, 32);
      s1 = WMMA_BF16(qa1, t, s1);
    }
    // ---- online softmax: row r+8*(lane>=16) lives in component r across a
    //      16-lane half; reduce over keys with xor-shuffles of width 16 ----
#pragma unroll
    for (int r = 0; r < 8; ++r) {
      float a = s0[r] * 0.125f;                      // * 64^-0.5
      float c = s1[r] * 0.125f;
      float mx = fmaxf(a, c);
      mx = fmaxf(mx, __shfl_xor(mx, 1, 16));
      mx = fmaxf(mx, __shfl_xor(mx, 2, 16));
      mx = fmaxf(mx, __shfl_xor(mx, 4, 16));
      mx = fmaxf(mx, __shfl_xor(mx, 8, 16));
      const float mnew = fmaxf(mrun[r], mx);
      const float al = __expf(mrun[r] - mnew);
      const float p0 = __expf(a - mnew);
      const float p1 = __expf(c - mnew);
      float rs = p0 + p1;
      rs += __shfl_xor(rs, 1, 16);
      rs += __shfl_xor(rs, 2, 16);
      rs += __shfl_xor(rs, 4, 16);
      rs += __shfl_xor(rs, 8, 16);
      lrun[r] = lrun[r] * al + rs;
      mrun[r] = mnew;
      o0[r] *= al; o1[r] *= al; o2[r] *= al; o3[r] *= al;
      const int row = rowoff + r;
      lds[wl][row * 32 + colbase]      = (bf16)p0;   // C-layout -> LDS [row][key]
      lds[wl][row * 32 + 16 + colbase] = (bf16)p1;
    }
    // ---- O += P[16x32] @ V[32x64] (V^T rows are key-contiguous) ----
    const v16bf pa = load_frag16(&lds[wl][0], 32, 0);
    v16bf vb = load_frag16(V + (size_t)kt, 2048, 0);
    o0 = WMMA_BF16(pa, vb, o0);
    vb = load_frag16(V + (size_t)16 * 2048 + kt, 2048, 0);
    o1 = WMMA_BF16(pa, vb, o1);
    vb = load_frag16(V + (size_t)32 * 2048 + kt, 2048, 0);
    o2 = WMMA_BF16(pa, vb, o2);
    vb = load_frag16(V + (size_t)48 * 2048 + kt, 2048, 0);
    o3 = WMMA_BF16(pa, vb, o3);
  }
  // softmax normalize * (s/sqrt(s-1)) * s^-0.5  == sqrt(2048/2047) / l
  const float fin = 1.000244200f;
  bf16* orow = ob + ((size_t)(b * 2048 + qt * 16 + rowoff)) * 1024 + h * 64 + colbase;
#pragma unroll
  for (int r = 0; r < 8; ++r) {
    const float inv = fin / lrun[r];
    bf16* p = orow + (size_t)r * 1024;
    p[0]  = (bf16)(o0[r] * inv);
    p[16] = (bf16)(o1[r] * inv);
    p[32] = (bf16)(o2[r] * inv);
    p[48] = (bf16)(o3[r] * inv);
  }
}

// ---------------------------------------------------------------------------
// Kernel 4: output projection. y[4096,1024] = Ab @ Wo^T / 32, f32 out.
// One wave computes a 32x64 block (8 WMMAs per k-step).
// ---------------------------------------------------------------------------
__global__ void __launch_bounds__(256) gemm_out_kernel(const bf16* __restrict__ ab,
                                                       const bf16* __restrict__ wob,
                                                       float* __restrict__ y) {
  const int wave = (blockIdx.x * 256 + threadIdx.x) >> 5;
  const int mt = wave >> 4;                          // 0..127 (32 rows each)
  const int nt = wave & 15;                          // 0..15  (64 cols each)
  const bf16* A0 = ab  + (size_t)mt * 32 * 1024;
  const bf16* A1 = A0 + 16 * 1024;
  const bf16* B  = wob + (size_t)nt * 64 * 1024;
  v8f acc[2][4] = {};
  for (int kk = 0; kk < 1024; kk += 32) {
    const v16bf a0 = load_frag16(A0, 1024, kk);
    const v16bf a1 = load_frag16(A1, 1024, kk);
#pragma unroll
    for (int j = 0; j < 4; ++j) {
      const v16bf bfr = load_frag16(B + (size_t)j * 16 * 1024, 1024, kk);
      acc[0][j] = WMMA_BF16(a0, bfr, acc[0][j]);
      acc[1][j] = WMMA_BF16(a1, bfr, acc[1][j]);
    }
  }
  const int lane = threadIdx.x & 31;
  const int m0   = mt * 32 + ((lane >> 4) << 3);
#pragma unroll
  for (int t = 0; t < 2; ++t) {
#pragma unroll
    for (int j = 0; j < 4; ++j) {
      const int col = nt * 64 + j * 16 + (lane & 15);
#pragma unroll
      for (int r = 0; r < 8; ++r)
        y[(size_t)(m0 + t * 16 + r) * 1024 + col] = acc[t][j][r] * 0.03125f;
    }
  }
}

// ---------------------------------------------------------------------------
extern "C" void kernel_launch(void* const* d_in, const int* in_sizes, int n_in,
                              void* d_out, int out_size, void* d_ws, size_t ws_size,
                              hipStream_t stream) {
  const float* x    = (const float*)d_in[0];   // [2,2048,1024]
  const float* wqkv = (const float*)d_in[1];   // [3072,1024]
  const float* wo   = (const float*)d_in[2];   // [1024,1024]
  float* out = (float*)d_out;                  // [2,2048,1024]

  bf16* ws = (bf16*)d_ws;
  bf16* xb    = ws;                            // 4096*1024
  bf16* wqkvb = xb    + (size_t)4096 * 1024;   // 3072*1024
  bf16* wob   = wqkvb + (size_t)3072 * 1024;   // 1024*1024
  bf16* qb    = wob   + (size_t)1024 * 1024;   // 32*2048*64
  bf16* kb    = qb    + (size_t)32 * 2048 * 64;
  bf16* vtb   = kb    + (size_t)32 * 2048 * 64;
  bf16* aout  = vtb   + (size_t)32 * 2048 * 64; // 4096*1024
  // total: ~48 MB of bf16 workspace

  cast_f32_bf16<<<(4096 * 1024) / (256 * 8), 256, 0, stream>>>(x, xb);
  cast_f32_bf16<<<(3072 * 1024) / (256 * 8), 256, 0, stream>>>(wqkv, wqkvb);
  cast_f32_bf16<<<(1024 * 1024) / (256 * 8), 256, 0, stream>>>(wo, wob);

  // 128 m-blocks * 48 n-blocks = 6144 waves, 8 waves per block
  gemm_qkv_kernel<<<6144 / 8, 256, 0, stream>>>(xb, wqkvb, qb, kb, vtb);

  // 32 (b,h) * 128 q-tiles = 4096 waves
  attn_kernel<<<4096 / 8, 256, 0, stream>>>(qb, kb, vtb, aout);

  // 128 m-blocks * 16 n-blocks = 2048 waves
  gemm_out_kernel<<<2048 / 8, 256, 0, stream>>>(aout, wob, out);
}
